// EpidemicSurrogateNet_85298050498696
// MI455X (gfx1250) — compile-verified, matrix operand order checked by reading, not run on hardware
//
#include <hip/hip_runtime.h>

typedef _Float16 half_t;
typedef __attribute__((ext_vector_type(16))) _Float16 v16h;
typedef __attribute__((ext_vector_type(8)))  _Float16 v8h;
typedef __attribute__((ext_vector_type(8)))  float    v8f;
typedef __attribute__((ext_vector_type(4)))  float    v4f;

#define B_SZ     4096
#define H_DIM    128
#define T_STEPS  160
#define WG_ROWS  16          // batch rows per workgroup (one 16-row M-tile)
#define HPAD     136         // padded LDS row stride in halves (272B, 16B aligned)

// d_ws layout (bytes)
#define OFF_B0    0u                      // Whh0 packed B-fragments (f16), 128KB
#define OFF_B1    131072u                 // Wih1 packed
#define OFF_B2    262144u                 // Whh1 packed
#define OFF_BIAS0 393216u                 // bih0+bhh0 (512 f32)
#define OFF_BIAS1 395264u                 // bih1+bhh1 (512 f32)
#define OFF_WIH0  397312u                 // Wih0 column (512 f32)
#define OFF_WD    399360u                 // Wd (128 f32)
#define OFF_BD    399872u                 // bd (1 f32)
#define OFF_H0    400384u                 // initial h0, f16 [4096][128]
#define OFF_H1    (OFF_H0 + 1048576u)
#define OFF_C0    (OFF_H1 + 1048576u)     // f32 [4096][128]
#define OFF_C1    (OFF_C0 + 2097152u)

#define WMMA_F16(a,b,c) \
  __builtin_amdgcn_wmma_f32_16x16x32_f16(false,(a),false,(b),(short)0,(c),false,false)

__device__ __forceinline__ float fsig(float x)  { return 1.0f / (1.0f + __expf(-x)); }
__device__ __forceinline__ float ftanh(float x) { float e = __expf(-2.0f * x); return (1.0f - e) / (1.0f + e); }

// ---------------------------------------------------------------------------
// Pack Whh0 / Wih1 / Whh1 (each [512][128] f32, gates = h @ W^T) into f16
// WMMA B-fragment layout. Tile = (kf: 32 K) x (nt: 16 N). Per tile 32 lanes x
// 16 halves contiguous (1KB). Assumed B layout = transpose-of-A layout:
// lane l -> N = 16*nt + (l&15); half hh -> VGPR v=hh/2, pos=hh&1,
// k_local = (v<4 ? 2v+pos : 16+2(v-4)+pos) + (l>=16 ? 8 : 0).
// ---------------------------------------------------------------------------
__global__ void pack_weights(const float* __restrict__ Whh0,
                             const float* __restrict__ Wih1,
                             const float* __restrict__ Whh1,
                             half_t* __restrict__ dst) {
  int tid = blockIdx.x * blockDim.x + threadIdx.x;
  if (tid >= 3 * 65536) return;
  int mat  = tid >> 16;
  int rem  = tid & 65535;
  int tile = rem >> 9;            // (nt*4 + kf), 0..127
  int lane = (rem >> 4) & 31;
  int hh   = rem & 15;
  int nt = tile >> 2, kf = tile & 3;
  int v = hh >> 1, pos = hh & 1;
  int kl = (v < 4 ? 2 * v + pos : 16 + 2 * (v - 4) + pos) + (lane >= 16 ? 8 : 0);
  int n = nt * 16 + (lane & 15);
  int k = kf * 32 + kl;
  const float* W = (mat == 0) ? Whh0 : ((mat == 1) ? Wih1 : Whh1);
  dst[tid] = (half_t)W[n * H_DIM + k];
}

__global__ void pack_small(const float* __restrict__ bih0, const float* __restrict__ bhh0,
                           const float* __restrict__ bih1, const float* __restrict__ bhh1,
                           const float* __restrict__ Wih0, const float* __restrict__ Wd,
                           const float* __restrict__ bd, char* __restrict__ ws) {
  int i = threadIdx.x;  // 512 threads, 1 block
  float* b0 = (float*)(ws + OFF_BIAS0);
  float* b1 = (float*)(ws + OFF_BIAS1);
  float* w0 = (float*)(ws + OFF_WIH0);
  float* wd = (float*)(ws + OFF_WD);
  float* bv = (float*)(ws + OFF_BD);
  if (i < 512) {
    b0[i] = bih0[i] + bhh0[i];
    b1[i] = bih1[i] + bhh1[i];
    w0[i] = Wih0[i];
  }
  if (i < 128) wd[i] = Wd[i];
  if (i == 0)  bv[0] = bd[0];
}

// ---------------------------------------------------------------------------
// Encoder: e=relu(x@We1^T); e=relu(e@We2^T); hc=e@We3^T -> initial h0,c0,h1,c1
// One block (128 thr) per batch row. Trivial one-time cost.
// ---------------------------------------------------------------------------
__global__ void encoder(const float* __restrict__ x,
                        const float* __restrict__ We1, const float* __restrict__ be1,
                        const float* __restrict__ We2, const float* __restrict__ be2,
                        const float* __restrict__ We3, const float* __restrict__ be3,
                        char* __restrict__ ws) {
  __shared__ float xs[4];
  __shared__ float e1[64];
  __shared__ float e2[128];
  int r = blockIdx.x, tid = threadIdx.x;
  if (tid < 4) xs[tid] = x[r * 4 + tid];
  __syncthreads();
  if (tid < 64) {
    float s = be1[tid];
    #pragma unroll
    for (int k = 0; k < 4; ++k) s += We1[tid * 4 + k] * xs[k];
    e1[tid] = fmaxf(s, 0.f);
  }
  __syncthreads();
  {
    float s = be2[tid];
    #pragma unroll 8
    for (int k = 0; k < 64; ++k) s += We2[tid * 64 + k] * e1[k];
    e2[tid] = fmaxf(s, 0.f);
  }
  __syncthreads();
  half_t* H0 = (half_t*)(ws + OFF_H0);
  half_t* H1 = (half_t*)(ws + OFF_H1);
  float*  C0 = (float*)(ws + OFF_C0);
  float*  C1 = (float*)(ws + OFF_C1);
  #pragma unroll
  for (int q = 0; q < 4; ++q) {
    int n = q * 128 + tid;
    float s = be3[n];
    #pragma unroll 8
    for (int k = 0; k < 128; ++k) s += We3[n * 128 + k] * e2[k];
    if      (q == 0) H0[r * 128 + tid] = (half_t)s;
    else if (q == 1) C0[r * 128 + tid] = s;
    else if (q == 2) H1[r * 128 + tid] = (half_t)s;
    else             C1[r * 128 + tid] = s;
  }
}

// ---------------------------------------------------------------------------
// Persistent LSTM rollout. 256 WGs x 8 wave32. WG owns one 16-row M-tile;
// wave w owns hidden columns [16w,16w+16). h-states in LDS (f16, padded rows),
// c-states in VGPR C-fragments. All gate GEMMs via v_wmma_f32_16x16x32_f16.
// Per-wave live set: A 32 + acc 32 + c 16 + B-in-flight + consts -> fits the
// 256-VGPR budget from __launch_bounds__(256,4) with slack for B prefetch.
// ---------------------------------------------------------------------------
__device__ __forceinline__ v16h load_afrag(const half_t* hs, int arow, int kc, int laneHi) {
  int base = arow * HPAD + laneHi * 8;                 // halves
  v8h lo = *(const v8h*)(hs + base + kc * 32);
  v8h hi = *(const v8h*)(hs + base + kc * 32 + 16);
  return __builtin_shufflevector(lo, hi, 0,1,2,3,4,5,6,7,8,9,10,11,12,13,14,15);
}

__global__ __launch_bounds__(256, 4)
void lstm_rollout(char* __restrict__ ws, float* __restrict__ out) {
  __shared__ __align__(16) half_t h0s[WG_ROWS * HPAD];
  __shared__ __align__(16) half_t h1s[WG_ROWS * HPAD];
  __shared__ __align__(16) float  predb[WG_ROWS];
  __shared__ float wds[128];

  const int tid  = threadIdx.x;
  const int w    = tid >> 5;        // wave id 0..7 -> j-chunk
  const int lane = tid & 31;
  const int jn   = lane & 15;
  const int lh   = lane >> 4;
  const int row0 = blockIdx.x * WG_ROWS;

  const v16h*  Bv  = (const v16h*)(ws + OFF_B0);   // 4096 v16h per matrix
  const float* b0g = (const float*)(ws + OFF_BIAS0);
  const float* b1g = (const float*)(ws + OFF_BIAS1);
  const float* w0g = (const float*)(ws + OFF_WIH0);
  const float* wdg = (const float*)(ws + OFF_WD);
  const float  bdv = *(const float*)(ws + OFF_BD);

  // hoisted per-lane gate constants: n(g) = g*128 + 16w + jn
  float bias0[4], bias1[4], wih0c[4];
  #pragma unroll
  for (int g = 0; g < 4; ++g) {
    int n = g * 128 + w * 16 + jn;
    bias0[g] = b0g[n];  bias1[g] = b1g[n];  wih0c[g] = w0g[n];
  }

  // stage initial h into LDS
  const half_t* H0g = (const half_t*)(ws + OFF_H0) + (size_t)row0 * H_DIM;
  const half_t* H1g = (const half_t*)(ws + OFF_H1) + (size_t)row0 * H_DIM;
  for (int i = tid; i < WG_ROWS * H_DIM; i += 256) {
    int r = i >> 7, c = i & 127;
    h0s[r * HPAD + c] = H0g[i];
    h1s[r * HPAD + c] = H1g[i];
  }
  if (tid < 128)     wds[tid]  = wdg[tid];
  if (tid < WG_ROWS) predb[tid] = 0.f;

  // c-states in C-fragment layout: element (m = r + 8*lh, n = 16w + jn)
  float c0f[8], c1f[8];
  const float* C0g = (const float*)(ws + OFF_C0);
  const float* C1g = (const float*)(ws + OFF_C1);
  #pragma unroll
  for (int r = 0; r < 8; ++r) {
    int m = row0 + r + 8 * lh;
    int n = w * 16 + jn;
    c0f[r] = C0g[m * H_DIM + n];
    c1f[r] = C1g[m * H_DIM + n];
  }
  __syncthreads();

  for (int t = 0; t < T_STEPS; ++t) {
    v8f acc[4];

    // ---------------- cell 0: gates0 = h0 @ Whh0^T ----------------
    {
      v16h a[4];
      #pragma unroll
      for (int kc = 0; kc < 4; ++kc) a[kc] = load_afrag(h0s, jn, kc, lh);
      #pragma unroll
      for (int g = 0; g < 4; ++g) {
        v8f c = {0.f,0.f,0.f,0.f,0.f,0.f,0.f,0.f};
        #pragma unroll
        for (int kf = 0; kf < 4; ++kf) {
          v16h bf = Bv[((g * 8 + w) * 4 + kf) * 32 + lane];
          c = WMMA_F16(a[kf], bf, c);
        }
        acc[g] = c;
      }
    }
    __syncthreads();   // all waves finished reading h0

    {
      float pr[8];
      {
        const v4f* pp = (const v4f*)(predb + lh * 8);
        v4f p0 = pp[0], p1 = pp[1];
        pr[0]=p0[0]; pr[1]=p0[1]; pr[2]=p0[2]; pr[3]=p0[3];
        pr[4]=p1[0]; pr[5]=p1[1]; pr[6]=p1[2]; pr[7]=p1[3];
      }
      #pragma unroll
      for (int r = 0; r < 8; ++r) {
        float ii = fsig (acc[0][r] + bias0[0] + pr[r] * wih0c[0]);
        float ff = fsig (acc[1][r] + bias0[1] + pr[r] * wih0c[1]);
        float gg = ftanh(acc[2][r] + bias0[2] + pr[r] * wih0c[2]);
        float oo = fsig (acc[3][r] + bias0[3] + pr[r] * wih0c[3]);
        float cn = ff * c0f[r] + ii * gg;
        c0f[r] = cn;
        float hn = oo * ftanh(cn);
        int rrow = r + 8 * lh;
        h0s[rrow * HPAD + w * 16 + jn] = (half_t)hn;
      }
    }
    __syncthreads();   // h0_new visible

    // ------------- cell 1: gates1 = h0_new @ Wih1^T + h1 @ Whh1^T -------------
    {
      // pass 1: A = h0_new, B = Wih1, acc = 0
      v16h a[4];
      #pragma unroll
      for (int kc = 0; kc < 4; ++kc) a[kc] = load_afrag(h0s, jn, kc, lh);
      #pragma unroll
      for (int g = 0; g < 4; ++g) {
        v8f c = {0.f,0.f,0.f,0.f,0.f,0.f,0.f,0.f};
        #pragma unroll
        for (int kf = 0; kf < 4; ++kf) {
          v16h bf = Bv[4096 + ((g * 8 + w) * 4 + kf) * 32 + lane];
          c = WMMA_F16(a[kf], bf, c);
        }
        acc[g] = c;
      }
      // pass 2: A = h1, B = Whh1, accumulate
      #pragma unroll
      for (int kc = 0; kc < 4; ++kc) a[kc] = load_afrag(h1s, jn, kc, lh);
      #pragma unroll
      for (int g = 0; g < 4; ++g) {
        v8f c = acc[g];
        #pragma unroll
        for (int kf = 0; kf < 4; ++kf) {
          v16h bf = Bv[8192 + ((g * 8 + w) * 4 + kf) * 32 + lane];
          c = WMMA_F16(a[kf], bf, c);
        }
        acc[g] = c;
      }
    }
    __syncthreads();   // all waves finished reading h1

    #pragma unroll
    for (int r = 0; r < 8; ++r) {
      float ii = fsig (acc[0][r] + bias1[0]);
      float ff = fsig (acc[1][r] + bias1[1]);
      float gg = ftanh(acc[2][r] + bias1[2]);
      float oo = fsig (acc[3][r] + bias1[3]);
      float cn = ff * c1f[r] + ii * gg;
      c1f[r] = cn;
      float hn = oo * ftanh(cn);
      int rrow = r + 8 * lh;
      h1s[rrow * HPAD + w * 16 + jn] = (half_t)hn;
    }
    __syncthreads();   // h1_new visible

    // ------------- decoder: pred = sigmoid(h1_new @ Wd^T + bd) -------------
    if (tid < WG_ROWS) {
      const half_t* hr = h1s + tid * HPAD;
      float a0 = 0.f, a1 = 0.f, a2 = 0.f, a3 = 0.f;
      #pragma unroll
      for (int j = 0; j < 128; j += 4) {
        a0 += (float)hr[j + 0] * wds[j + 0];
        a1 += (float)hr[j + 1] * wds[j + 1];
        a2 += (float)hr[j + 2] * wds[j + 2];
        a3 += (float)hr[j + 3] * wds[j + 3];
      }
      float p = fsig(a0 + a1 + a2 + a3 + bdv);
      predb[tid] = p;
      out[(size_t)(row0 + tid) * T_STEPS + t] = p;
    }
    __syncthreads();   // pred visible for next step's cell 0
  }
}

// ---------------------------------------------------------------------------
extern "C" void kernel_launch(void* const* d_in, const int* in_sizes, int n_in,
                              void* d_out, int out_size, void* d_ws, size_t ws_size,
                              hipStream_t stream) {
  (void)in_sizes; (void)n_in; (void)out_size; (void)ws_size;
  const float* x    = (const float*)d_in[0];
  const float* We1  = (const float*)d_in[1];
  const float* be1  = (const float*)d_in[2];
  const float* We2  = (const float*)d_in[3];
  const float* be2  = (const float*)d_in[4];
  const float* We3  = (const float*)d_in[5];
  const float* be3  = (const float*)d_in[6];
  const float* Wih0 = (const float*)d_in[7];
  const float* Whh0 = (const float*)d_in[8];
  const float* bih0 = (const float*)d_in[9];
  const float* bhh0 = (const float*)d_in[10];
  const float* Wih1 = (const float*)d_in[11];
  const float* Whh1 = (const float*)d_in[12];
  const float* bih1 = (const float*)d_in[13];
  const float* bhh1 = (const float*)d_in[14];
  const float* Wd   = (const float*)d_in[15];
  const float* bd   = (const float*)d_in[16];
  char* ws = (char*)d_ws;

  pack_weights<<<768, 256, 0, stream>>>(Whh0, Wih1, Whh1, (half_t*)(ws + OFF_B0));
  pack_small  <<<1, 512, 0, stream>>>(bih0, bhh0, bih1, bhh1, Wih0, Wd, bd, ws);
  encoder     <<<B_SZ, 128, 0, stream>>>(x, We1, be1, We2, be2, We3, be3, ws);
  lstm_rollout<<<B_SZ / WG_ROWS, 256, 0, stream>>>(ws, (float*)d_out);
}
